// CRFLoss_89137751261300
// MI455X (gfx1250) — matrix operand order repaired and checked by reading, not compiled, not run on hardware
//
#include <hip/hip_runtime.h>
#include <hip/hip_bf16.h>
#include <math.h>

typedef __attribute__((ext_vector_type(16))) _Float16 v16h;
typedef __attribute__((ext_vector_type(8)))  float    v8f;

#define B_    256
#define L_    1024
#define T_    128
#define BTILE 16

#define LOG2E_F 1.4426950408889634f
#define LN2_F   0.6931471805599453f

// ---------------------------------------------------------------------------
// Forward-score kernel: one workgroup (8 wave32) per 16-batch tile.
// Recursion in exp-domain:  p_new[k] = f[k] + M + log( V(16x128) . expT(128x128) )
// where V[m][j] = exp(p[m][j]-M[m]), expT[j][k] = exp(trans[j][k]).
// Per step: 8 waves x 4 v_wmma_f32_16x16x32_f16 cover the 16x128 output.
// ---------------------------------------------------------------------------
__global__ __launch_bounds__(256)
void crf_forward_kernel(const float* __restrict__ trans,
                        const float* __restrict__ feats,
                        const unsigned char* __restrict__ mask,
                        float* __restrict__ out)
{
    __shared__ _Float16 sB[T_ * T_];                         // [n][j] = exp(trans[j][n]) (B-frag K contiguous)
    __shared__ _Float16 sV[BTILE * T_];                      // [m][j] A-matrix (exp domain)
    __shared__ float    sP[BTILE * T_];                      // partition state (f32)
    __shared__ __attribute__((aligned(16)))
             unsigned char sMaskT[L_ * BTILE];               // mask transposed: [t][b]
    __shared__ float    sM[BTILE];                           // per-row max

    const int tid  = threadIdx.x;
    const int b0   = blockIdx.x * BTILE;
    const int wv   = tid >> 5;            // wave id 0..7 -> N-tile [16*wv, 16*wv+16)
    const int lane = tid & 31;
    const int mrow = lane & 15;           // A-frag row
    const int hi   = lane >> 4;           // lane half
    const int ncol = (lane & 15) + 16 * wv;

    // exp(transitions) -> LDS, transposed [n][j]
    for (int idx = tid; idx < T_ * T_; idx += 256) {
        int j = idx >> 7, n = idx & 127;
        sB[n * T_ + j] = (_Float16)__builtin_amdgcn_exp2f(trans[idx] * LOG2E_F);
    }
    // mask tile -> LDS, transposed [t][b] (one ds_load_b64 per lane per step later)
    for (int idx = tid; idx < BTILE * L_; idx += 256) {
        int b = idx >> 10, t = idx & (L_ - 1);
        sMaskT[t * BTILE + b] = mask[(size_t)(b0 + b) * L_ + t];
    }
    // partition init: p = feats[:, 0, :]
    for (int idx = tid; idx < BTILE * T_; idx += 256) {
        int r = idx >> 7, k = idx & 127;
        sP[idx] = feats[(size_t)(b0 + r) * (L_ * T_) + k];
    }
    __syncthreads();

    const int rrow = tid >> 4;            // 0..15: partition row owned for reductions
    const int rc   = tid & 15;            // 16 threads per row, 8 contiguous cols each

    for (int t = 1; t < L_; ++t) {
        // Prefetch next step's feats tile (16 rows x 512B).
        if (t + 1 < L_ && tid < 64) {
            int pr = tid >> 2, pc = (tid & 3) * 32;
            __builtin_prefetch(&feats[(size_t)(b0 + pr) * (L_ * T_) + (size_t)(t + 1) * T_ + pc], 0, 1);
        }

        // (1) row max via half-wave shuffles (no LDS, no barrier)
        float pv[8];
        float m = -3.4e38f;
        #pragma unroll
        for (int i = 0; i < 8; ++i) {
            pv[i] = sP[rrow * T_ + rc * 8 + i];
            m = fmaxf(m, pv[i]);
        }
        m = fmaxf(m, __shfl_xor(m, 1, 16));
        m = fmaxf(m, __shfl_xor(m, 2, 16));
        m = fmaxf(m, __shfl_xor(m, 4, 16));
        m = fmaxf(m, __shfl_xor(m, 8, 16));
        if (rc == 0) sM[rrow] = m;

        // (2) V = exp(p - M) in f16 (values in (0,1], f16-safe; raw v_exp_f32)
        #pragma unroll
        for (int i = 0; i < 8; ++i)
            sV[rrow * T_ + rc * 8 + i] =
                (_Float16)__builtin_amdgcn_exp2f((pv[i] - m) * LOG2E_F);
        __syncthreads();   // barrier 1: sV / sM ready

        // Hoist step-t data loads so they overlap the WMMAs.
        unsigned long long mk =
            *(const unsigned long long*)&sMaskT[t * BTILE + hi * 8];
        float fv[8];
        #pragma unroll
        for (int r = 0; r < 8; ++r) {
            int gb = b0 + r + hi * 8;
            fv[r] = feats[(size_t)gb * (L_ * T_) + (size_t)t * T_ + ncol];
        }

        // (3) W tile = V x expT via 4 chained WMMAs (K = 128)
        v8f acc = {};
        #pragma unroll
        for (int kk = 0; kk < 4; ++kk) {
            const int kbase = kk * 32;
            v16h a, bf;
            #pragma unroll
            for (int i = 0; i < 8; ++i) {   // A frag (16-bit A 16x32 ISA layout)
                int K = kbase + ((i < 4) ? 0 : 16) + hi * 8 + (i & 3) * 2;
                a[2 * i]     = sV[mrow * T_ + K];
                a[2 * i + 1] = sV[mrow * T_ + K + 1];
            }
            #pragma unroll
            for (int q = 0; q < 16; ++q) {  // B frag: K = 8*hi + q + (q>=8 ? 8 : 0) (+ kbase), col = ncol
                bf[q] = sB[ncol * T_ + kbase + hi * 8 + q + ((q >= 8) ? 8 : 0)];
            }
            acc = __builtin_amdgcn_wmma_f32_16x16x32_f16(false, a, false, bf,
                                                         (short)0, acc, false, false);
        }

        // (4) branchless masked partition update.
        //     C layout: VGPR r -> M = r + 8*hi, N = ncol.
        #pragma unroll
        for (int r = 0; r < 8; ++r) {
            int bm  = r + hi * 8;
            float pn  = fv[r] + sM[bm] + __builtin_amdgcn_logf(acc[r]) * LN2_F;
            float old = sP[bm * T_ + ncol];
            bool  mkb = ((mk >> (8 * r)) & 0xFFull) != 0ull;
            sP[bm * T_ + ncol] = mkb ? pn : old;
        }
        __syncthreads();   // barrier 2: sP consistent for next step
    }

    // Final: sum over rows of logsumexp(partition[row, :]) via half-wave shuffles
    float pv[8];
    float m = -3.4e38f;
    #pragma unroll
    for (int i = 0; i < 8; ++i) {
        pv[i] = sP[rrow * T_ + rc * 8 + i];
        m = fmaxf(m, pv[i]);
    }
    m = fmaxf(m, __shfl_xor(m, 1, 16));
    m = fmaxf(m, __shfl_xor(m, 2, 16));
    m = fmaxf(m, __shfl_xor(m, 4, 16));
    m = fmaxf(m, __shfl_xor(m, 8, 16));
    float s = 0.f;
    #pragma unroll
    for (int i = 0; i < 8; ++i)
        s += __builtin_amdgcn_exp2f((pv[i] - m) * LOG2E_F);
    s += __shfl_xor(s, 1, 16);
    s += __shfl_xor(s, 2, 16);
    s += __shfl_xor(s, 4, 16);
    s += __shfl_xor(s, 8, 16);
    if (rc == 0) sM[rrow] = m + __builtin_amdgcn_logf(s) * LN2_F;
    __syncthreads();
    if (tid == 0) {
        float tot = 0.f;
        #pragma unroll
        for (int r = 0; r < BTILE; ++r) tot += sM[r];
        atomicAdd(out, tot);
    }
}

// ---------------------------------------------------------------------------
// Gold-score kernel: gather unary + binary scores, subtract from output.
// ---------------------------------------------------------------------------
__global__ __launch_bounds__(256)
void crf_gold_kernel(const float* __restrict__ trans,
                     const float* __restrict__ feats,
                     const unsigned char* __restrict__ mask,
                     const int* __restrict__ tags,
                     float* __restrict__ out)
{
    __shared__ float red[256];
    const int b   = blockIdx.x;
    const int tid = threadIdx.x;
    float s = 0.f;
    for (int l = tid; l < L_; l += 256) {
        if (mask[(size_t)b * L_ + l]) {
            int tg = tags[(size_t)b * L_ + l];
            s += feats[(size_t)b * (L_ * T_) + (size_t)l * T_ + tg];   // unary
            if (l >= 1) {
                int tp = tags[(size_t)b * L_ + l - 1];
                s += trans[tp * T_ + tg];                              // binary
            }
        }
    }
    red[tid] = s;
    __syncthreads();
    for (int off = 128; off > 0; off >>= 1) {
        if (tid < off) red[tid] += red[tid + off];
        __syncthreads();
    }
    if (tid == 0) atomicAdd(out, -red[0]);
}

extern "C" void kernel_launch(void* const* d_in, const int* in_sizes, int n_in,
                              void* d_out, int out_size, void* d_ws, size_t ws_size,
                              hipStream_t stream) {
    const float*         trans = (const float*)d_in[0];          // (T,T) f32
    const float*         feats = (const float*)d_in[1];          // (B,L,T) f32
    const unsigned char* mask  = (const unsigned char*)d_in[2];  // (B,L) bool (1 byte)
    const int*           tags  = (const int*)d_in[3];            // (B,L) int

    float* out = (float*)d_out;                                  // scalar f32
    hipMemsetAsync(out, 0, sizeof(float), stream);               // graph-capture safe

    crf_forward_kernel<<<B_ / BTILE, 256, 0, stream>>>(trans, feats, mask, out);
    crf_gold_kernel<<<B_, 256, 0, stream>>>(trans, feats, mask, tags, out);
}
